// MHA_15066745275082
// MI455X (gfx1250) — matrix operand order repaired and checked by reading, not compiled, not run on hardware
//
#include <hip/hip_runtime.h>
#include <hip/hip_bf16.h>
#include <cstdint>
#include <cstddef>

// Problem constants (match reference)
#define BB   4
#define CC   512
#define KKK  2048
#define HH   8
#define DHH  128
#define MTOT (BB * KKK)   // 8192 rows (b*K + kpos)

typedef __attribute__((ext_vector_type(16))) __bf16 bf16x16;
typedef __attribute__((ext_vector_type(8)))  __bf16 bf16x8;
typedef __attribute__((ext_vector_type(8)))  float  f32x8;

static __device__ __forceinline__ bf16x16 join8(bf16x8 lo, bf16x8 hi) {
  union { bf16x16 v; bf16x8 h[2]; } u;
  u.h[0] = lo; u.h[1] = hi;
  return u.v;
}

// A-fragment (16x32 bf16): lane holds row=l16, contiguous K
// {kk + half*16 .. +15} (matches load_bfrag). One 32B load.
static __device__ __forceinline__ bf16x16 load_afrag(const __bf16* rowp, int kk, int half) {
  return *(const bf16x16*)(rowp + kk + half * 16);
}

// B-fragment (32x16 bf16) sourced from a K-major row (weight row n):
// lane holds col=l16, contiguous K {kk + half*16 .. +15}. One 32B load.
static __device__ __forceinline__ bf16x16 load_bfrag(const __bf16* rowp, int kk, int half) {
  return *(const bf16x16*)(rowp + kk + half * 16);
}

#define WMMA_BF16(a, b, c) \
  __builtin_amdgcn_wmma_f32_16x16x32_bf16(false, (a), false, (b), (short)0, (c), false, false)

// Wave-level GEMM mainloop: one wave computes a 16 x (16*NF) tile of
// D = A(MxKd, K-major bf16) * W(NxKd, K-major bf16)^T, f32 accumulate.
// All NF+1 fragment loads of a k-step are batched into live registers before
// the WMMA burst so the waits amortize instead of serializing load->wmma.
template <int NF>
static __device__ __forceinline__ void gemm_tile(const __bf16* __restrict__ A, int lda,
                                                 const __bf16* __restrict__ W, int ldw,
                                                 int Kd, int m0, int n0, f32x8 (&acc)[NF]) {
  const int lane = threadIdx.x & 31;
  const int l16 = lane & 15, half = lane >> 4;
  const __bf16* arow = A + (size_t)(m0 + l16) * lda;
  for (int kk = 0; kk < Kd; kk += 32) {
    bf16x16 af = load_afrag(arow, kk, half);
    bf16x16 bfr[NF];
#pragma unroll
    for (int f = 0; f < NF; ++f)
      bfr[f] = load_bfrag(W + (size_t)(n0 + f * 16 + l16) * ldw, kk, half);
#pragma unroll
    for (int f = 0; f < NF; ++f)
      acc[f] = WMMA_BF16(af, bfr[f], acc[f]);
  }
}

// ---------------------------------------------------------------- utilities
__global__ void cvt_bf16_kernel(const float* __restrict__ in, __bf16* __restrict__ out, int n) {
  int i = blockIdx.x * blockDim.x + threadIdx.x;
  if (i < n) out[i] = (__bf16)in[i];
}

// LayerNorm over C with transpose: x(B,C,K) -> xn,res (M=b*K+kpos, C) bf16
__global__ __launch_bounds__(256) void ln_kernel(const float* __restrict__ x,
                                                 const float* __restrict__ g,
                                                 const float* __restrict__ bta,
                                                 __bf16* __restrict__ xn,
                                                 __bf16* __restrict__ res) {
  const int row = blockIdx.x;            // b*K + kpos
  const int b = row >> 11, kpos = row & (KKK - 1);
  const float* xb = x + (size_t)b * CC * KKK + kpos;
  __shared__ float red[2][8];
  const int c0 = threadIdx.x, c1 = threadIdx.x + 256;
  float v0 = xb[(size_t)c0 * KKK];
  float v1 = xb[(size_t)c1 * KKK];
  float s = v0 + v1, s2 = v0 * v0 + v1 * v1;
#pragma unroll
  for (int off = 16; off; off >>= 1) {
    s += __shfl_xor(s, off, 32);
    s2 += __shfl_xor(s2, off, 32);
  }
  const int wv = threadIdx.x >> 5;
  if ((threadIdx.x & 31) == 0) { red[0][wv] = s; red[1][wv] = s2; }
  __syncthreads();
  s = 0.f; s2 = 0.f;
#pragma unroll
  for (int i = 0; i < 8; ++i) { s += red[0][i]; s2 += red[1][i]; }
  const float mean = s * (1.f / CC);
  const float var = s2 * (1.f / CC) - mean * mean;
  const float rstd = rsqrtf(var + 1e-6f);
  const size_t o = (size_t)row * CC;
  xn[o + c0] = (__bf16)((v0 - mean) * rstd * g[c0] + bta[c0]);
  xn[o + c1] = (__bf16)((v1 - mean) * rstd * g[c1] + bta[c1]);
  res[o + c0] = (__bf16)v0;
  res[o + c1] = (__bf16)v1;
}

// ------------------------------------------------- Q/K projection: N = 1024
// out layout (B,H,K,DH) bf16; `scale` folds 1/sqrt(DH) into Q.
__global__ __launch_bounds__(128) void proj_qk_kernel(const __bf16* __restrict__ A,
                                                      const __bf16* __restrict__ W,
                                                      __bf16* __restrict__ out, float scale) {
  const int wave = threadIdx.x >> 5;
  const int lane = threadIdx.x & 31;
  const int l16 = lane & 15, half = lane >> 4;
  const int m0 = blockIdx.x * 64 + wave * 16;
  const int n0 = blockIdx.y * 128;
  f32x8 acc[8] = {};
  gemm_tile<8>(A, CC, W, CC, CC, m0, n0, acc);
  const int b = m0 >> 11;
#pragma unroll
  for (int f = 0; f < 8; ++f) {
    const int n = n0 + f * 16 + l16;
    const int h = n >> 7, d = n & (DHH - 1);
#pragma unroll
    for (int r = 0; r < 8; ++r) {
      const int kpos = (m0 & (KKK - 1)) + r + half * 8;
      out[(((size_t)b * HH + h) * KKK + kpos) * DHH + d] = (__bf16)(acc[f][r] * scale);
    }
  }
}

// ------------------------------------------------- V projection: N = 4096
// out layout V^T: (B,H,C,K) bf16 so the attention B-fragment is contiguous.
__global__ __launch_bounds__(128) void proj_v_kernel(const __bf16* __restrict__ A,
                                                     const __bf16* __restrict__ W,
                                                     __bf16* __restrict__ vt) {
  const int wave = threadIdx.x >> 5;
  const int lane = threadIdx.x & 31;
  const int l16 = lane & 15, half = lane >> 4;
  const int m0 = blockIdx.x * 64 + wave * 16;
  const int n0 = blockIdx.y * 128;
  f32x8 acc[8] = {};
  gemm_tile<8>(A, CC, W, CC, CC, m0, n0, acc);
  const int b = m0 >> 11;
#pragma unroll
  for (int f = 0; f < 8; ++f) {
    const int n = n0 + f * 16 + l16;
    const int h = n >> 9, c = n & (CC - 1);
#pragma unroll
    for (int r = 0; r < 8; ++r) {
      const int kpos = (m0 & (KKK - 1)) + r + half * 8;
      vt[(((size_t)b * HH + h) * CC + c) * KKK + kpos] = (__bf16)acc[f][r];
    }
  }
}

// ------------------------------------------------------- flash attention
// block = 4 waves, 16 q-rows; each wave owns 128 output columns. Each wave
// redundantly computes the 16x64 score tile (keeps waves independent: no
// block barrier in the hot loop), runs online softmax in the D-layout, and
// round-trips P through a private LDS tile to get A-layout fragments.
// LDS ops are in-order per wave (ISA 7.3), so the cross-lane store->load
// only needs a compiler scheduling fence (wave_barrier), not s_wait_dscnt;
// V-tile global loads are hoisted above the LDS round-trip to overlap it.
__global__ __launch_bounds__(128) void attn_kernel(const __bf16* __restrict__ qg,
                                                   const __bf16* __restrict__ kg,
                                                   const __bf16* __restrict__ vtg,
                                                   __bf16* __restrict__ rout) {
  __shared__ alignas(32) __bf16 Plds[4][16][64];
  const int bh = blockIdx.x;                // 0..31
  const int q0 = blockIdx.y * 16;           // q tile base
  const int wave = threadIdx.x >> 5;
  const int lane = threadIdx.x & 31;
  const int l16 = lane & 15, half = lane >> 4;
  const int c0 = wave * 128;

  const __bf16* Q  = qg  + (size_t)bh * KKK * DHH;
  const __bf16* Km = kg  + (size_t)bh * KKK * DHH;
  const __bf16* Vt = vtg + (size_t)bh * CC * KKK;

  // Q A-fragments for the whole kernel (DH = 128 -> 4 frags)
  bf16x16 qf[4];
  const __bf16* qrow = Q + (size_t)(q0 + l16) * DHH;
#pragma unroll
  for (int d = 0; d < 4; ++d) qf[d] = load_afrag(qrow, d * 32, half);

  f32x8 acc[8] = {};
  float m[8], l[8];
#pragma unroll
  for (int r = 0; r < 8; ++r) { m[r] = -1e30f; l[r] = 0.f; }

  for (int kt = 0; kt < KKK; kt += 64) {
    // ---- scores S = (Q*scale) . K^T for 16 q-rows x 64 k-cols
    f32x8 s[4] = {};
#pragma unroll
    for (int d = 0; d < 4; ++d) {
      bf16x16 kf[4];
#pragma unroll
      for (int f = 0; f < 4; ++f)
        kf[f] = load_bfrag(Km + (size_t)(kt + f * 16 + l16) * DHH, d * 32, half);
#pragma unroll
      for (int f = 0; f < 4; ++f)
        s[f] = WMMA_BF16(qf[d], kf[f], s[f]);
    }
    // ---- online softmax (row r lives in slot r; lanes 0-15 rows 0-7,
    //      lanes 16-31 rows 8-15; reduce over the 16-lane N group)
    float alpha[8];
#pragma unroll
    for (int r = 0; r < 8; ++r) {
      float v = fmaxf(fmaxf(s[0][r], s[1][r]), fmaxf(s[2][r], s[3][r]));
#pragma unroll
      for (int off = 8; off; off >>= 1) v = fmaxf(v, __shfl_xor(v, off, 32));
      const float mn = fmaxf(m[r], v);
      alpha[r] = __expf(m[r] - mn);
      m[r] = mn;
    }
    float rowsum[8];
#pragma unroll
    for (int r = 0; r < 8; ++r) rowsum[r] = 0.f;
#pragma unroll
    for (int f = 0; f < 4; ++f) {
#pragma unroll
      for (int r = 0; r < 8; ++r) {
        const float p = __expf(s[f][r] - m[r]);
        s[f][r] = p;
        rowsum[r] += p;
      }
    }
#pragma unroll
    for (int r = 0; r < 8; ++r) {
      float v = rowsum[r];
#pragma unroll
      for (int off = 8; off; off >>= 1) v += __shfl_xor(v, off, 32);
      l[r] = l[r] * alpha[r] + v;
    }
#pragma unroll
    for (int f = 0; f < 8; ++f)
#pragma unroll
      for (int r = 0; r < 8; ++r) acc[f][r] *= alpha[r];

    // ---- prefetch first V fragment group (independent of the LDS round-trip)
    bf16x16 vf[8];
#pragma unroll
    for (int f = 0; f < 8; ++f)
      vf[f] = load_bfrag(Vt + (size_t)(c0 + f * 16 + l16) * KKK + kt, 0, half);

    // ---- D-layout -> LDS (row-major 16x64 bf16, per-wave private tile)
#pragma unroll
    for (int f = 0; f < 4; ++f)
#pragma unroll
      for (int r = 0; r < 8; ++r)
        Plds[wave][r + half * 8][f * 16 + l16] = (__bf16)s[f][r];
    __builtin_amdgcn_wave_barrier();   // stores before loads; LDS in-order per wave

    // ---- O += P . V   (B-fragments from transposed V: contiguous rows)
    bf16x16 pf = load_afrag(&Plds[wave][l16][0], 0, half);
#pragma unroll
    for (int f = 0; f < 8; ++f)
      acc[f] = WMMA_BF16(pf, vf[f], acc[f]);
#pragma unroll
    for (int f = 0; f < 8; ++f)
      vf[f] = load_bfrag(Vt + (size_t)(c0 + f * 16 + l16) * KKK + kt, 32, half);
    pf = load_afrag(&Plds[wave][l16][0], 32, half);
#pragma unroll
    for (int f = 0; f < 8; ++f)
      acc[f] = WMMA_BF16(pf, vf[f], acc[f]);
    __builtin_amdgcn_wave_barrier();   // this tile's P loads before next tile's stores
  }

  // ---- normalize and write r (M, H*C) bf16
  const int b = bh >> 3, h = bh & 7;
  float inv[8];
#pragma unroll
  for (int r = 0; r < 8; ++r) inv[r] = 1.0f / l[r];
#pragma unroll
  for (int f = 0; f < 8; ++f) {
    const int col = c0 + f * 16 + l16;
#pragma unroll
    for (int r = 0; r < 8; ++r) {
      const int row = q0 + r + half * 8;
      rout[(size_t)(b * KKK + row) * (HH * CC) + h * CC + col] = (__bf16)(acc[f][r] * inv[r]);
    }
  }
}

// ------------------------------------------------- r @ Ww^T : 8192x4096 -> 512
__global__ __launch_bounds__(128) void ww_kernel(const __bf16* __restrict__ A,
                                                 const __bf16* __restrict__ W,
                                                 __bf16* __restrict__ y1) {
  const int wave = threadIdx.x >> 5;
  const int lane = threadIdx.x & 31;
  const int l16 = lane & 15, half = lane >> 4;
  const int m0 = blockIdx.x * 64 + wave * 16;
  const int n0 = blockIdx.y * 128;
  f32x8 acc[8] = {};
  gemm_tile<8>(A, HH * CC, W, HH * CC, HH * CC, m0, n0, acc);
#pragma unroll
  for (int f = 0; f < 8; ++f) {
    const int n = n0 + f * 16 + l16;
#pragma unroll
    for (int r = 0; r < 8; ++r)
      y1[(size_t)(m0 + r + half * 8) * CC + n] = (__bf16)acc[f][r];
  }
}

// ------------------- [y1, residual] @ Wa^T, store f32 transposed to (B,C,K)
__global__ __launch_bounds__(128) void wa_kernel(const __bf16* __restrict__ y1,
                                                 const __bf16* __restrict__ res,
                                                 const __bf16* __restrict__ Wa,
                                                 float* __restrict__ out) {
  const int wave = threadIdx.x >> 5;
  const int lane = threadIdx.x & 31;
  const int l16 = lane & 15, half = lane >> 4;
  const int m0 = blockIdx.x * 64 + wave * 16;
  const int n0 = blockIdx.y * 128;
  f32x8 acc[8] = {};
  gemm_tile<8>(y1, CC, Wa, 2 * CC, CC, m0, n0, acc);          // first half of concat
  gemm_tile<8>(res, CC, Wa + CC, 2 * CC, CC, m0, n0, acc);    // residual half
  const int b = m0 >> 11;
#pragma unroll
  for (int f = 0; f < 8; ++f) {
    const int n = n0 + f * 16 + l16;
#pragma unroll
    for (int r = 0; r < 8; ++r) {
      const int kpos = (m0 & (KKK - 1)) + r + half * 8;
      out[((size_t)b * CC + n) * KKK + kpos] = acc[f][r];
    }
  }
}

// ----------------------------------------------------------------- launcher
extern "C" void kernel_launch(void* const* d_in, const int* in_sizes, int n_in,
                              void* d_out, int out_size, void* d_ws, size_t ws_size,
                              hipStream_t stream) {
  (void)in_sizes; (void)n_in; (void)out_size; (void)ws_size;
  const float* x  = (const float*)d_in[0];
  const float* g  = (const float*)d_in[1];
  const float* bt = (const float*)d_in[2];
  const float* Wq = (const float*)d_in[3];
  const float* Wk = (const float*)d_in[4];
  const float* Wv = (const float*)d_in[5];
  const float* Ww = (const float*)d_in[6];
  const float* Wa = (const float*)d_in[7];
  float* out = (float*)d_out;

  char* base = (char*)d_ws;
  size_t off = 0;
  auto carve = [&](size_t elems) -> __bf16* {
    off = (off + 255) & ~(size_t)255;
    __bf16* p = (__bf16*)(base + off);
    off += elems * sizeof(__bf16);
    return p;
  };
  __bf16* xnb  = carve((size_t)MTOT * CC);          // LN output
  __bf16* resb = carve((size_t)MTOT * CC);          // residual (pre-LN, transposed)
  __bf16* wqb  = carve((size_t)1024 * 512);
  __bf16* wkb  = carve((size_t)1024 * 512);
  __bf16* wvb  = carve((size_t)4096 * 512);
  __bf16* wwb  = carve((size_t)512 * 4096);
  __bf16* wab  = carve((size_t)512 * 1024);
  __bf16* qb   = carve((size_t)BB * HH * KKK * DHH);  // (B,H,K,DH), pre-scaled
  __bf16* kb   = carve((size_t)BB * HH * KKK * DHH);  // (B,H,K,DH)
  __bf16* vtb  = carve((size_t)BB * HH * CC * KKK);   // (B,H,C,K)  V transposed
  __bf16* rb   = carve((size_t)MTOT * HH * CC);       // attention result (M, 4096)
  __bf16* y1b  = carve((size_t)MTOT * CC);            // r @ Ww^T

  // 1) weights fp32 -> bf16
  cvt_bf16_kernel<<<(1024 * 512 + 255) / 256, 256, 0, stream>>>(Wq, wqb, 1024 * 512);
  cvt_bf16_kernel<<<(1024 * 512 + 255) / 256, 256, 0, stream>>>(Wk, wkb, 1024 * 512);
  cvt_bf16_kernel<<<(4096 * 512 + 255) / 256, 256, 0, stream>>>(Wv, wvb, 4096 * 512);
  cvt_bf16_kernel<<<(512 * 4096 + 255) / 256, 256, 0, stream>>>(Ww, wwb, 512 * 4096);
  cvt_bf16_kernel<<<(512 * 1024 + 255) / 256, 256, 0, stream>>>(Wa, wab, 512 * 1024);

  // 2) LayerNorm + transpose (+ bf16 residual)
  ln_kernel<<<MTOT, 256, 0, stream>>>(x, g, bt, xnb, resb);

  // 3) projections (Q gets 1/sqrt(DH) folded in)
  const float qscale = 0.088388347648318447f;  // 1/sqrt(128)
  proj_qk_kernel<<<dim3(MTOT / 64, 1024 / 128), 128, 0, stream>>>(xnb, wqb, qb, qscale);
  proj_qk_kernel<<<dim3(MTOT / 64, 1024 / 128), 128, 0, stream>>>(xnb, wkb, kb, 1.0f);
  proj_v_kernel<<<dim3(MTOT / 64, 4096 / 128), 128, 0, stream>>>(xnb, wvb, vtb);

  // 4) flash attention per (b,h) x q-tile
  attn_kernel<<<dim3(BB * HH, KKK / 16), 128, 0, stream>>>(qb, kb, vtb, rb);

  // 5) output projection + fused concat/Wa + transpose to (B,C,K) f32
  ww_kernel<<<dim3(MTOT / 64, 512 / 128), 128, 0, stream>>>(rb, wwb, y1b);
  wa_kernel<<<dim3(MTOT / 64, 512 / 128), 128, 0, stream>>>(y1b, resb, wab, out);
}
